// MAEAttention_68642167325258
// MI455X (gfx1250) — matrix operand order repaired and checked by reading, not compiled, not run on hardware
//
#include <hip/hip_runtime.h>
#include <hip/hip_bf16.h>

typedef _Float16 h8  __attribute__((ext_vector_type(8)));
typedef _Float16 h16 __attribute__((ext_vector_type(16)));
typedef float    f8v __attribute__((ext_vector_type(8)));
typedef int      v4i __attribute__((ext_vector_type(4)));

#if defined(__gfx1250__) && __has_builtin(__builtin_amdgcn_global_load_async_to_lds_b128) && \
    __has_builtin(__builtin_amdgcn_s_wait_asynccnt)
#define HAVE_ASYNC 1
#else
#define HAVE_ASYNC 0
#endif

#if HAVE_ASYNC
// 16-byte async DMA: global -> LDS, tracked by ASYNCcnt.
// Builtin signature (per hipcc diagnostic): (int4 AS1*, int4 AS3*, imm, imm)
__device__ __forceinline__ void async_copy16(const void* g, void* l) {
  __builtin_amdgcn_global_load_async_to_lds_b128(
      (__attribute__((address_space(1))) v4i*)g,
      (__attribute__((address_space(3))) v4i*)l, 0, 0);
}
#define ASYNC_WAIT(n) __builtin_amdgcn_s_wait_asynccnt(n)
#else
__device__ __forceinline__ void async_copy16(const void* g, void* l) {
  *(h8*)l = *(const h8*)g;  // synchronous fallback
}
#define ASYNC_WAIT(n) ((void)0)
#endif

__device__ __forceinline__ f8v wmma16x16x32(h16 a, h16 b, f8v c) {
  // D = A(16x32 f16) * B(32x16 f16) + C(16x16 f32)
  return __builtin_amdgcn_wmma_f32_16x16x32_f16(false, a, false, b, (short)0, c,
                                                false, false);
}

// ---------------------------------------------------------------------------
// fp32 -> f16 conversion (one-time, so GEMM tiles can be DMA'd without cvt)
// ---------------------------------------------------------------------------
__global__ __launch_bounds__(256) void cvt_f32_to_f16(
    const float* __restrict__ src, _Float16* __restrict__ dst, int n) {
  const int i4 = (blockIdx.x * 256 + threadIdx.x) * 4;
  if (i4 + 3 < n) {
    const float4 v = *(const float4*)(src + i4);
    dst[i4 + 0] = (_Float16)v.x;
    dst[i4 + 1] = (_Float16)v.y;
    dst[i4 + 2] = (_Float16)v.z;
    dst[i4 + 3] = (_Float16)v.w;
  } else {
    for (int i = i4; i < n; ++i) dst[i] = (_Float16)src[i];
  }
}

// ---------------------------------------------------------------------------
// Materialize biasF[h][i][j] = rpb_table[rp_index[i][j]][h] (12*197*197 f32,
// 1.86 MB -> L2 resident). Turns the attention-side double gather into a
// single coalesced stream.
// ---------------------------------------------------------------------------
__global__ __launch_bounds__(256) void build_bias(
    const float* __restrict__ rpb, const int* __restrict__ rpi,
    float* __restrict__ biasF) {
  constexpr int NN = 197 * 197;
  const int idx = blockIdx.x * 256 + threadIdx.x;
  if (idx < 12 * NN) {
    const int h = idx / NN, ij = idx - h * NN;
    biasF[idx] = rpb[rpi[ij] * 12 + h];
  }
}

// ---------------------------------------------------------------------------
// Tiled GEMM (all-f16 operands): C[M,N] = A[M,K] * W[N,K]^T + bias[N]
// Block tile 128x128, 256 threads = 8 wave32, K fixed at 768 (24 steps of 32).
// Double-buffered LDS tiles filled by async global->LDS b128 DMA; each thread
// issues 4 DMAs per K-step, waits asynccnt<=4 so the next tile stays in flight.
// ---------------------------------------------------------------------------
template <typename TOUT>
__global__ __launch_bounds__(256) void gemm_f16_bias_wT(
    const _Float16* __restrict__ A, const _Float16* __restrict__ W,
    const float* __restrict__ bias, TOUT* __restrict__ C, int M, int N) {
  constexpr int K = 768;
  constexpr int NK = K / 32;
  __shared__ __attribute__((aligned(64))) _Float16 As[2][128 * 32];  // [m][k]
  __shared__ __attribute__((aligned(64))) _Float16 Bs[2][128 * 32];  // [n][k]

  const int tid = threadIdx.x;
  const int lane = tid & 31;
  const int wv = tid >> 5;
  const int lhi = lane >> 4;
  const int llo = lane & 15;
  const int m0 = blockIdx.y * 128;
  const int n0 = blockIdx.x * 128;
  const int wm = wv & 3;   // 4 M slots x 32 rows
  const int wn = wv >> 2;  // 2 N slots x 64 cols

  // issue one 128x32 f16 tile (A and B) as 2+2 b128 DMAs per thread
  auto load_tile = [&](int k0, int buf) {
#pragma unroll
    for (int it = 0; it < 2; ++it) {
      const int c = tid + 256 * it;        // 512 chunks of 8 halves
      const int r = c >> 2, kc = (c & 3) * 8;
      int m = m0 + r;
      if (m > M - 1) m = M - 1;            // clamp; stores guarded later
      async_copy16(&A[(size_t)m * K + k0 + kc], &As[buf][r * 32 + kc]);
      async_copy16(&W[(size_t)(n0 + r) * K + k0 + kc], &Bs[buf][r * 32 + kc]);
    }
  };

  const f8v vzero = {0.f, 0.f, 0.f, 0.f, 0.f, 0.f, 0.f, 0.f};
  f8v acc[2][4];
#pragma unroll
  for (int mt = 0; mt < 2; ++mt)
#pragma unroll
    for (int nt = 0; nt < 4; ++nt) acc[mt][nt] = vzero;

  load_tile(0, 0);
  for (int i = 0; i < NK; ++i) {
    const int buf = i & 1;
    if (i + 1 < NK) {
      load_tile((i + 1) * 32, buf ^ 1);
      ASYNC_WAIT(4);  // oldest 4 (tile i) complete; tile i+1 stays in flight
    } else {
      ASYNC_WAIT(0);
    }
    __syncthreads();  // tile i visible to all waves

    h16 a[2];
#pragma unroll
    for (int mt = 0; mt < 2; ++mt) {
      const int ml = 32 * wm + 16 * mt + llo;
      const int kb = 16 * lhi;
      h8 c1 = *(const h8*)&As[buf][ml * 32 + kb];
      h8 c2 = *(const h8*)&As[buf][ml * 32 + kb + 8];
#pragma unroll
      for (int e = 0; e < 8; ++e) { a[mt][e] = c1[e]; a[mt][e + 8] = c2[e]; }
    }
#pragma unroll
    for (int nt = 0; nt < 4; ++nt) {
      const int nl = 64 * wn + 16 * nt + llo;
      h16 bm = *(const h16*)&Bs[buf][nl * 32 + 16 * lhi];
#pragma unroll
      for (int mt = 0; mt < 2; ++mt)
        acc[mt][nt] = wmma16x16x32(a[mt], bm, acc[mt][nt]);
    }
    __syncthreads();  // done reading buf before it is refilled
  }

#pragma unroll
  for (int nt = 0; nt < 4; ++nt) {
    const int n = n0 + 64 * wn + 16 * nt + llo;
    const float bv = bias[n];
#pragma unroll
    for (int mt = 0; mt < 2; ++mt) {
#pragma unroll
      for (int r = 0; r < 8; ++r) {
        const int m = m0 + 32 * wm + 16 * mt + r + 8 * lhi;
        if (m < M) C[(size_t)m * N + n] = (TOUT)(acc[mt][nt][r] + bv);
      }
    }
  }
}

// ---------------------------------------------------------------------------
// Fused attention: one workgroup per (b, h); 13 wave32 (416 threads), each
// wave owns one 16-row query block of the 208-padded 197 tokens.
// S = scale*Q*K^T + bias, masked softmax in registers, O = P*V.
// ---------------------------------------------------------------------------
__global__ __launch_bounds__(416) void attn_softmax_pv(
    const _Float16* __restrict__ qkv,        // [B*197, 2304] f16 (q|k|v)
    const float* __restrict__ biasF,         // [12, 197, 197] precomputed
    const unsigned char* __restrict__ mask,  // [B, 197] bool
    _Float16* __restrict__ attout) {         // [B*197, 768] f16
  constexpr int NT = 197, NP = 208, KP = 224, HD = 64, C3 = 2304;
  __shared__ __attribute__((aligned(64))) _Float16 Ks[NP * HD];    // [tok][d]
  __shared__ __attribute__((aligned(64))) _Float16 Vt[HD * KP];    // [d][tok]
  __shared__ __attribute__((aligned(64))) _Float16 Stg[13 * 256];  // staging

  const int h = blockIdx.x, b = blockIdx.y;
  const int tid = threadIdx.x, lane = tid & 31, wv = tid >> 5;
  const int lhi = lane >> 4, llo = lane & 15;
  const size_t rowbase = (size_t)b * NT * C3;
  const float* hbias = biasF + (size_t)h * NT * NT;

  // K tile: 208*64 halves = 1664 b128 chunks = exactly 4 per thread (async DMA)
#pragma unroll
  for (int it = 0; it < 4; ++it) {
    const int c = tid + 416 * it;
    const int t = c >> 3, kc = (c & 7) * 8;
    const int tc = t < NT ? t : NT - 1;  // clamped rows are masked later
    async_copy16(&qkv[rowbase + (size_t)tc * C3 + 768 + h * 64 + kc],
                 &Ks[t * 64 + kc]);
  }
  // V^T tile (transposed, zero-padded) -- manual; overlaps the async K loads
  for (int idx = tid; idx < HD * KP; idx += 416) {
    const int d = idx / KP, t = idx - d * KP;
    _Float16 v = (_Float16)0.f;
    if (t < NT) v = qkv[rowbase + (size_t)t * C3 + 1536 + h * 64 + d];
    Vt[idx] = v;
  }
  ASYNC_WAIT(0);
  __syncthreads();

  // Q block loaded straight from global in WMMA A-layout (2 K-steps over hd=64)
  h16 aq[2];
  {
    int tok = 16 * wv + llo;
    if (tok > NT - 1) tok = NT - 1;
    const _Float16* qrow = qkv + rowbase + (size_t)tok * C3 + h * 64;
#pragma unroll
    for (int kk = 0; kk < 2; ++kk) {
      const int d1 = 32 * kk + 16 * lhi;
      h8 c1 = *(const h8*)(qrow + d1);
      h8 c2 = *(const h8*)(qrow + d1 + 8);
#pragma unroll
      for (int e = 0; e < 8; ++e) { aq[kk][e] = c1[e]; aq[kk][e + 8] = c2[e]; }
    }
  }

  const f8v vzero = {0.f, 0.f, 0.f, 0.f, 0.f, 0.f, 0.f, 0.f};
  f8v s[13];
#pragma unroll
  for (int nt = 0; nt < 13; ++nt) s[nt] = vzero;

  // S = Q * K^T  (13 key tiles x 2 K-steps)
#pragma unroll
  for (int nt = 0; nt < 13; ++nt) {
#pragma unroll
    for (int kk = 0; kk < 2; ++kk) {
      h16 bm = *(const h16*)&Ks[(16 * nt + llo) * HD + 32 * kk + 16 * lhi];
      s[nt] = wmma16x16x32(aq[kk], bm, s[nt]);
    }
  }

  // scale + precomputed bias (coalesced: consecutive lanes -> consecutive j)
  const float scale = 0.125f;  // 64^-0.5
#pragma unroll
  for (int nt = 0; nt < 13; ++nt) {
    const int j = 16 * nt + llo;
    const int jc = j < NT ? j : NT - 1;
    const bool jok = (j < NT) && (mask[b * NT + j] != 0);
#pragma unroll
    for (int r = 0; r < 8; ++r) {
      int i = 16 * wv + r + 8 * lhi;
      if (i > NT - 1) i = NT - 1;
      const float v = s[nt][r] * scale + hbias[i * NT + jc];
      s[nt][r] = jok ? v : -1e30f;
    }
  }

  // register softmax; row r lives on VGPR r across the 16 lanes of a half-wave
#pragma unroll
  for (int r = 0; r < 8; ++r) {
    float mx = -1e30f;
#pragma unroll
    for (int nt = 0; nt < 13; ++nt) mx = fmaxf(mx, s[nt][r]);
#pragma unroll
    for (int off = 1; off < 16; off <<= 1) mx = fmaxf(mx, __shfl_xor(mx, off, 32));
    float sum = 0.f;
#pragma unroll
    for (int nt = 0; nt < 13; ++nt) {
      const float e = __expf(s[nt][r] - mx);
      s[nt][r] = e;
      sum += e;
    }
#pragma unroll
    for (int off = 1; off < 16; off <<= 1) sum += __shfl_xor(sum, off, 32);
    const float inv = 1.f / sum;
#pragma unroll
    for (int nt = 0; nt < 13; ++nt) s[nt][r] *= inv;
  }

  // O = P * V: re-layout P (C-layout -> A-layout) through per-wave LDS staging.
  // Per-wave DS ops are in-order, so store->load->store needs no barrier.
  _Float16* stg = &Stg[wv * 256];
  f8v o[4];
#pragma unroll
  for (int dt = 0; dt < 4; ++dt) o[dt] = vzero;

#pragma unroll
  for (int ks = 0; ks < 7; ++ks) {
    h16 ap;
#pragma unroll
    for (int r = 0; r < 8; ++r)
      stg[(r + 8 * lhi) * 16 + llo] = (_Float16)s[2 * ks][r];
    {
      h8 c1 = *(const h8*)&stg[llo * 16 + 8 * lhi];
#pragma unroll
      for (int e = 0; e < 8; ++e) ap[e] = c1[e];
    }
#pragma unroll
    for (int r = 0; r < 8; ++r) {
      _Float16 pv = (_Float16)0.f;
      if (2 * ks + 1 < 13) pv = (_Float16)s[(2 * ks + 1 < 13) ? (2 * ks + 1) : 0][r];
      stg[(r + 8 * lhi) * 16 + llo] = pv;
    }
    {
      h8 c2 = *(const h8*)&stg[llo * 16 + 8 * lhi];
#pragma unroll
      for (int e = 0; e < 8; ++e) ap[e + 8] = c2[e];
    }
#pragma unroll
    for (int dt = 0; dt < 4; ++dt) {
      h8 v1 = *(const h8*)&Vt[(16 * dt + llo) * KP + 32 * ks + 8 * lhi];
      h8 v2 = *(const h8*)&Vt[(16 * dt + llo) * KP + 32 * ks + 8 * lhi + 16];
      h16 bm;
#pragma unroll
      for (int e = 0; e < 8; ++e) { bm[e] = v1[e]; bm[e + 8] = v2[e]; }
      o[dt] = wmma16x16x32(ap, bm, o[dt]);
    }
  }

  // write O in [B*197, 768] layout (col = h*64 + d), f16 for the proj GEMM
#pragma unroll
  for (int dt = 0; dt < 4; ++dt) {
#pragma unroll
    for (int r = 0; r < 8; ++r) {
      const int m = 16 * wv + r + 8 * lhi;
      if (m < NT)
        attout[(size_t)(b * NT + m) * 768 + h * 64 + 16 * dt + llo] =
            (_Float16)o[dt][r];
    }
  }
}

// ---------------------------------------------------------------------------
extern "C" void kernel_launch(void* const* d_in, const int* in_sizes, int n_in,
                              void* d_out, int out_size, void* d_ws,
                              size_t ws_size, hipStream_t stream) {
  (void)n_in; (void)out_size; (void)ws_size;
  const float* x      = (const float*)d_in[0];
  const float* qkv_w  = (const float*)d_in[1];
  const float* qkv_b  = (const float*)d_in[2];
  const float* proj_w = (const float*)d_in[3];
  const float* proj_b = (const float*)d_in[4];
  const float* rpb    = (const float*)d_in[5];
  const int*   rpi    = (const int*)d_in[6];
  const unsigned char* msk = (const unsigned char*)d_in[7];

  const int NT = 197, DIM = 768, HEADS = 12;
  const int Bsz = in_sizes[0] / (NT * DIM);  // 64
  const int M = Bsz * NT;                    // 12608

  // workspace layout:
  //   x16     [M, 768]    f16  19.4 MB
  //   wqkv16  [2304, 768] f16   3.5 MB
  //   wproj16 [768, 768]  f16   1.2 MB
  //   qkv16   [M, 2304]   f16  58.1 MB
  //   att16   [M, 768]    f16  19.4 MB
  //   biasF   [12,197,197]f32   1.9 MB
  _Float16* x16     = (_Float16*)d_ws;
  _Float16* wqkv16  = x16 + (size_t)M * DIM;
  _Float16* wproj16 = wqkv16 + (size_t)3 * DIM * DIM;
  _Float16* qkv16   = wproj16 + (size_t)DIM * DIM;
  _Float16* att16   = qkv16 + (size_t)M * 3 * DIM;
  float*    biasF   = (float*)(att16 + (size_t)M * DIM);

  const int nx = M * DIM, nwq = 3 * DIM * DIM, nwp = DIM * DIM;
  cvt_f32_to_f16<<<(nx / 4 + 255) / 256, 256, 0, stream>>>(x, x16, nx);
  cvt_f32_to_f16<<<(nwq / 4 + 255) / 256, 256, 0, stream>>>(qkv_w, wqkv16, nwq);
  cvt_f32_to_f16<<<(nwp / 4 + 255) / 256, 256, 0, stream>>>(proj_w, wproj16, nwp);

  const int nb = HEADS * NT * NT;
  build_bias<<<(nb + 255) / 256, 256, 0, stream>>>(rpb, rpi, biasF);

  // 1) qkv = x @ qkv_w^T + qkv_b   (f16 in, f16 out, async DMA tiles)
  gemm_f16_bias_wT<_Float16>
      <<<dim3(3 * DIM / 128, (M + 127) / 128), 256, 0, stream>>>(
          x16, wqkv16, qkv_b, qkv16, M, 3 * DIM);

  // 2) fused attention per (b, h)
  attn_softmax_pv<<<dim3(HEADS, Bsz), 416, 0, stream>>>(qkv16, biasF, msk,
                                                        att16);

  // 3) out = att @ proj_w^T + proj_b  (f16 in, fp32 out)
  gemm_f16_bias_wT<float>
      <<<dim3(DIM / 128, (M + 127) / 128), 256, 0, stream>>>(
          att16, wproj16, proj_b, (float*)d_out, M, DIM);
}
